// ScaledDotProductAttention_67757404061808
// MI455X (gfx1250) — matrix operand order repaired and checked by reading, not compiled, not run on hardware
//
#include <hip/hip_runtime.h>

typedef _Float16 half_t;
typedef __attribute__((ext_vector_type(16))) _Float16 v16h;
typedef __attribute__((ext_vector_type(8)))  _Float16 v8h;
typedef __attribute__((ext_vector_type(8)))  float    v8f;

#define S_LEN 2048
#define DHEAD 64
#define KT    64
#define NW    8

__device__ __forceinline__ v16h cat8(v8h a, v8h b) {
  return __builtin_shufflevector(a, b, 0,1,2,3,4,5,6,7,8,9,10,11,12,13,14,15);
}

__device__ __forceinline__ v8f wmma_f16(v16h a, v16h b, v8f c) {
  return __builtin_amdgcn_wmma_f32_16x16x32_f16(false, a, false, b, (short)0, c,
                                                false, false);
}

// Workgroup barrier that publishes LDS only: does NOT wait for in-flight
// global loads (unlike __syncthreads' full fence), so double-buffered tile
// loads stay outstanding across the barrier.
__device__ __forceinline__ void block_sync_lds() {
  asm volatile("s_wait_dscnt 0\n\t"
               "s_barrier_signal -1\n\t"
               "s_barrier_wait -1" ::: "memory");
}

// xor-butterfly lane-select constants for permlane16 (4 bits per lane).
__host__ __device__ constexpr unsigned psel_lo(int m) {
  unsigned s = 0;
  for (int i = 0; i < 8; ++i) s |= (unsigned)((i ^ m) & 0xF) << (4 * i);
  return s;
}
__host__ __device__ constexpr unsigned psel_hi(int m) {
  unsigned s = 0;
  for (int i = 8; i < 16; ++i) s |= (unsigned)((i ^ m) & 0xF) << (4 * (i - 8));
  return s;
}

// max with the value held by (lane ^ m) within each 16-lane row.
template <int M>
__device__ __forceinline__ float xor16_max(float v) {
#if __has_builtin(__builtin_amdgcn_permlane16)
  unsigned u = __float_as_uint(v);
  unsigned p = __builtin_amdgcn_permlane16(u, u, psel_lo(M), psel_hi(M),
                                           false, false);
  return fmaxf(v, __uint_as_float(p));
#else
  return fmaxf(v, __shfl_xor(v, M, 32));
#endif
}

__device__ __forceinline__ v8h cvt8(v8f f) {
  return __builtin_convertvector(f, v8h);
}

// Flash attention with ALiBi + causal mask, f16 WMMA (16x16x32), f32 accum.
// One 256-thread block (8 wave32) handles 128 query rows of one (b,h).
// 64-key tiles, LDS double-buffered with a DS-only barrier so next-tile
// global loads remain in flight across the sync and retire after compute.
__global__ __launch_bounds__(256)
void alibi_flash_attn(const float* __restrict__ Q,
                      const float* __restrict__ K,
                      const float* __restrict__ V,
                      const unsigned char* __restrict__ AM,
                      float* __restrict__ O)
{
  __shared__ half_t Ksh[2][KT][DHEAD];     // K tiles, row-major (key, d)    16 KB
  __shared__ half_t Vt[2][DHEAD][KT];      // V tiles, transposed (d, key)   16 KB
  __shared__ half_t Psh[NW][16][KT];       // per-wave P re-layout scratch   16 KB

  const int tid  = threadIdx.x;
  const int lane = tid & 31;
  const int w    = __builtin_amdgcn_readfirstlane(tid >> 5);  // scalar wave id
  const int ln   = lane & 15;
  const int hi   = lane >> 4;              // 0: lanes 0-15, 1: lanes 16-31

  const int bh   = blockIdx.x;             // 0..31  (b*16 + h)
  const int b    = bh >> 4;
  const int h    = bh & 15;
  const int qblk = blockIdx.y;             // 0..15

  // scalar: first q row of this wave -> skip/diag branches become s_cbranch
  const int qbase = __builtin_amdgcn_readfirstlane(qblk * 128 + w * 16);

  const float slope = exp2f(-0.5f * (float)(h + 1));   // 2^(-8/16*(h+1))
  const float L2E   = 1.44269504088896340736f;

  const size_t hoff = (size_t)bh * S_LEN * DHEAD;
  const float* qp = Q + hoff;
  const float* kp = K + hoff;
  const float* vp = V + hoff;
  const unsigned char* mp = AM + (size_t)b * S_LEN;

  // ---- Q tile -> WMMA A layout (f16), 1/sqrt(D) folded in ----
  v16h a_q[2];
  {
    const float* qr = qp + (size_t)(qbase + ln) * DHEAD;
    #pragma unroll
    for (int c = 0; c < 2; ++c) {
      v8h part[2];
      #pragma unroll
      for (int g = 0; g < 2; ++g) {
        const int db = c * 32 + hi * 8 + g * 16;
        float4 f0 = *(const float4*)(qr + db);
        float4 f1 = *(const float4*)(qr + db + 4);
        v8f fv;
        fv[0]=f0.x; fv[1]=f0.y; fv[2]=f0.z; fv[3]=f0.w;
        fv[4]=f1.x; fv[5]=f1.y; fv[6]=f1.z; fv[7]=f1.w;
        fv *= 0.125f;
        part[g] = cvt8(fv);
      }
      a_q[c] = cat8(part[0], part[1]);
    }
  }

  // Constant all-ones B operand: D = P x 1 puts the P row-sum in every lane.
  v16h ones_h;
  #pragma unroll
  for (int e = 0; e < 16; ++e) ones_h[e] = (half_t)1.0f;

  // O accumulators in WMMA C layout: VGPR r <-> row (r + hi*8), lane ln <-> col.
  v8f o_acc[4];
  v8f o_l;                                  // running softmax denominators
  float m_run[8];
  #pragma unroll
  for (int r = 0; r < 8; ++r) { m_run[r] = -3.0e38f; o_l[r] = 0.0f; }
  #pragma unroll
  for (int s4 = 0; s4 < 4; ++s4)
    #pragma unroll
    for (int r = 0; r < 8; ++r) o_acc[s4][r] = 0.0f;

  // ---- staging thread mapping (fixed per thread) ----
  const int key   = tid >> 2;              // K: key row 0..63
  const int dsg   = (tid & 3) * 16;        // K: d segment 0,16,32,48
  const int dcolV = tid & 63;              // V: d column (lanes consecutive)
  const int ksegV = (tid >> 6) * 16;       // V: key segment 0,16,32,48

  float kf[16], vf[16];                    // in-flight tile registers

  // loads of tile starting at k0 into registers (+prefetch one tile further)
  auto load_tile = [&](int k0) {
    const float* ks = kp + (size_t)(k0 + key) * DHEAD + dsg;
    #pragma unroll
    for (int g = 0; g < 2; ++g) {
      float4 f0 = *(const float4*)(ks + g*8);
      float4 f1 = *(const float4*)(ks + g*8 + 4);
      kf[g*8+0]=f0.x; kf[g*8+1]=f0.y; kf[g*8+2]=f0.z; kf[g*8+3]=f0.w;
      kf[g*8+4]=f1.x; kf[g*8+5]=f1.y; kf[g*8+6]=f1.z; kf[g*8+7]=f1.w;
    }
    const float* vs = vp + (size_t)(k0 + ksegV) * DHEAD + dcolV;
    #pragma unroll
    for (int j = 0; j < 16; ++j) vf[j] = vs[(size_t)j * DHEAD];
    if (k0 + 2 * KT <= S_LEN) {
      __builtin_prefetch(ks + KT * DHEAD, 0, 3);
      __builtin_prefetch(vs + KT * DHEAD, 0, 3);
    }
  };

  // convert + store in-flight registers into LDS buffer bufi
  auto store_tile = [&](int bufi) {
    #pragma unroll
    for (int g = 0; g < 2; ++g) {
      v8f fk, fv;
      #pragma unroll
      for (int j = 0; j < 8; ++j) { fk[j] = kf[g*8+j]; fv[j] = vf[g*8+j]; }
      *(v8h*)&Ksh[bufi][key][dsg + g*8]    = cvt8(fk);
      *(v8h*)&Vt[bufi][dcolV][ksegV + g*8] = cvt8(fv);
    }
  };

  const int nkt = qblk * 2 + 2;            // 64-key tiles up to block's last row

  // prologue: tile 0 into buffer 0
  load_tile(0);
  store_tile(0);

  for (int kt = 0; kt < nkt; ++kt) {
    const int k0  = kt * KT;
    const int cur = kt & 1;
    const bool have_next = (kt + 1 < nkt);
    const bool active    = (k0 <= qbase + 15);   // scalar branch

    // padding-mask bytes for THIS tile: issued first so the later loadcnt
    // wait for them does not cover the (subsequently issued) K/V tile loads.
    int mb[4];
    if (active) {
      #pragma unroll
      for (int sb = 0; sb < 4; ++sb) mb[sb] = mp[k0 + sb * 16 + ln];
    }

    if (have_next) load_tile(k0 + KT);     // issue global loads pre-barrier

    block_sync_lds();                      // buffer `cur` staged (DS-only wait)

    if (active) {
      // ---- S = Q K^T : 4 key-subtiles x 2 D-chunks = 8 WMMAs ----
      v8f s_t[4];
      #pragma unroll
      for (int sb = 0; sb < 4; ++sb) {
        const int keyl = sb * 16 + ln;
        v8f acc = {};
        #pragma unroll
        for (int c = 0; c < 2; ++c) {
          v8h blo = *(const v8h*)&Ksh[cur][keyl][c*32 + hi*8];
          v8h bhi = *(const v8h*)&Ksh[cur][keyl][c*32 + 16 + hi*8];
          acc = wmma_f16(a_q[c], cat8(blo, bhi), acc);
        }
        s_t[sb] = acc;
      }

      // ---- bias + masks: cheap path off-diagonal, full path on the single
      //      diagonal tile per wave (scalar branch) ----
      const bool diag = (k0 + KT > qbase);
      #pragma unroll
      for (int sb = 0; sb < 4; ++sb) {
        const int k_abs = k0 + sb * 16 + ln;
        float bias = slope * (float)(k_abs - (S_LEN - 1));
        bias = (mb[sb] != 0) ? bias : -1.0e30f;      // padding mask folded in
        if (diag) {
          #pragma unroll
          for (int r = 0; r < 8; ++r) {
            const int q_abs = qbase + r + hi * 8;
            s_t[sb][r] = (k_abs <= q_abs) ? (s_t[sb][r] + bias) : -1.0e30f;
          }
        } else {
          #pragma unroll
          for (int r = 0; r < 8; ++r) s_t[sb][r] += bias;
        }
      }

      // ---- online softmax: row max across 16-lane halves (VALU butterfly) ----
      float alpha[8], mL[8];
      #pragma unroll
      for (int r = 0; r < 8; ++r) {
        float mv = fmaxf(fmaxf(s_t[0][r], s_t[1][r]),
                         fmaxf(s_t[2][r], s_t[3][r]));
        mv = xor16_max<1>(mv);
        mv = xor16_max<2>(mv);
        mv = xor16_max<4>(mv);
        mv = xor16_max<8>(mv);
        const float m_new = fmaxf(m_run[r], mv);
        alpha[r] = exp2f((m_run[r] - m_new) * L2E);
        m_run[r] = m_new;
        mL[r]    = m_new * L2E;
      }

      // ---- P = exp(S - m) in f16 -> LDS; rescale running accumulators ----
      #pragma unroll
      for (int sb = 0; sb < 4; ++sb) {
        #pragma unroll
        for (int r = 0; r < 8; ++r) {
          const float p = exp2f(fmaf(s_t[sb][r], L2E, -mL[r]));
          Psh[w][r + hi*8][sb*16 + ln] = (half_t)p;   // C-layout -> LDS
        }
      }
      #pragma unroll
      for (int r = 0; r < 8; ++r) {
        o_l[r] *= alpha[r];
        #pragma unroll
        for (int s4 = 0; s4 < 4; ++s4) o_acc[s4][r] *= alpha[r];
      }

      // per-wave LDS scratch: our stores must land before the A-layout re-read
      asm volatile("s_wait_dscnt 0" ::: "memory");

      // ---- O += P V (8 WMMAs) and l += P x 1 (2 WMMAs) ----
      #pragma unroll
      for (int c = 0; c < 2; ++c) {
        v8h plo = *(const v8h*)&Psh[w][ln][c*32 + hi*8];
        v8h phi = *(const v8h*)&Psh[w][ln][c*32 + 16 + hi*8];
        const v16h a_p = cat8(plo, phi);
        #pragma unroll
        for (int s4 = 0; s4 < 4; ++s4) {
          const int dcol = s4 * 16 + ln;
          v8h vlo = *(const v8h*)&Vt[cur][dcol][c*32 + hi*8];
          v8h vhi = *(const v8h*)&Vt[cur][dcol][c*32 + 16 + hi*8];
          o_acc[s4] = wmma_f16(a_p, cat8(vlo, vhi), o_acc[s4]);
        }
        o_l = wmma_f16(a_p, ones_h, o_l);   // row sums, broadcast to all lanes
      }
    }

    // stage tile kt+1 into the other buffer; safe post-barrier (all waves are
    // done reading it), next iteration's barrier publishes it. The loadcnt
    // waits for kf/vf land here, after compute.
    if (have_next) store_tile(cur ^ 1);
  }

  // ---- epilogue: normalize by row sum, store f32 ----
  #pragma unroll
  for (int r = 0; r < 8; ++r) {
    const float inv = (o_l[r] > 0.0f) ? (1.0f / o_l[r]) : 0.0f;
    const int q_abs = qbase + r + hi * 8;
    float* orow = O + hoff + (size_t)q_abs * DHEAD;
    #pragma unroll
    for (int s4 = 0; s4 < 4; ++s4) {
      orow[s4 * 16 + ln] = o_acc[s4][r] * inv;
    }
  }
}

extern "C" void kernel_launch(void* const* d_in, const int* in_sizes, int n_in,
                              void* d_out, int out_size, void* d_ws, size_t ws_size,
                              hipStream_t stream) {
  const float* Q = (const float*)d_in[0];
  const float* K = (const float*)d_in[1];
  const float* V = (const float*)d_in[2];
  const unsigned char* AM = (const unsigned char*)d_in[3];
  float* O = (float*)d_out;
  (void)in_sizes; (void)n_in; (void)out_size; (void)d_ws; (void)ws_size;
  dim3 grid(32, 16);   // (b*h, q-block of 128 rows)
  dim3 block(256);     // 8 wave32
  alibi_flash_attn<<<grid, block, 0, stream>>>(Q, K, V, AM, O);
}
